// BiLSTM_9749575762681
// MI455X (gfx1250) — matrix operand order repaired and checked by reading, not compiled, not run on hardware
//
#include <hip/hip_runtime.h>

// Problem dims
#define S_LEN   256
#define BATCH   32
#define EMB     300
#define EMB_PAD 320
#define HID     256
#define IN1     512          // 2H input to layer 1
#define G4      1024         // 4*H gate width
#define TAGS    32
#define NROW    (S_LEN*BATCH)   // 8192 rows (s*32+b)

typedef __attribute__((ext_vector_type(16))) __bf16 v16bf;
typedef __attribute__((ext_vector_type(8)))  __bf16 v8bf;
typedef __attribute__((ext_vector_type(8)))  float  v8f;
typedef __attribute__((ext_vector_type(4)))  int    v4i;

#if defined(__has_builtin)
#if __has_builtin(__builtin_amdgcn_global_load_async_to_lds_b128)
#define ASYNC_LDS_OK 1
#endif
#endif

__device__ __forceinline__ v8f wmma_bf16(v16bf a, v16bf b, v8f c) {
  // v_wmma_f32_16x16x32_bf16
  return __builtin_amdgcn_wmma_f32_16x16x32_bf16(false, a, false, b, (short)0, c, false, false);
}

// A-fragment (16x32, row-major source): lane<16 -> K base 0, lane>=16 -> K base 8.
__device__ __forceinline__ v16bf load_a_frag(const __bf16* __restrict__ A, int lda,
                                             int row0, int k0, int lane) {
  int m  = lane & 15;
  int kb = (lane < 16) ? 0 : 8;
  const __bf16* p = A + (size_t)(row0 + m) * lda + k0 + kb;
  v8bf lo = *(const v8bf*)(p);
  v8bf hi = *(const v8bf*)(p + 16);
  v16bf r;
#pragma unroll
  for (int e = 0; e < 8; ++e) { r[e] = lo[e]; r[e + 8] = hi[e]; }
  return r;
}

// B-fragment (32x16 KxN) from W stored as [N][K] row-major: lane<16 -> K base 0, lane>=16 -> K base 8.
__device__ __forceinline__ v16bf load_b_frag(const __bf16* __restrict__ W, int ldw,
                                             int n0, int k0, int lane) {
  int n  = lane & 15;
  int kb = (lane < 16) ? 0 : 8;
  const __bf16* p = W + (size_t)(n0 + n) * ldw + k0 + kb;
  v8bf lo = *(const v8bf*)(p);
  v8bf hi = *(const v8bf*)(p + 16);
  v16bf r;
#pragma unroll
  for (int e = 0; e < 8; ++e) { r[e] = lo[e]; r[e + 8] = hi[e]; }
  return r;
}

__device__ __forceinline__ float sigm(float x) { return 1.0f / (1.0f + __expf(-x)); }

// Stream one step's gate block (32 rows x 1024 f32 = 128KB) into LDS.
// Async path uses ASYNCcnt (decoupled from LOADcnt/DScnt waits in the WMMA loop).
__device__ __forceinline__ void fetch_gates(const float* __restrict__ gsrc,
                                            float* glds, int tid) {
#pragma unroll
  for (int i = 0; i < 32; ++i) {
    int e4 = (i * 256 + tid) * 4;     // float index, 16B granules
#ifdef ASYNC_LDS_OK
    __builtin_amdgcn_global_load_async_to_lds_b128((v4i*)(gsrc + e4),
                                                   (v4i*)(glds + e4), 0, 0);
#else
    *(float4*)(glds + e4) = *(const float4*)(gsrc + e4);
#endif
  }
}

__device__ __forceinline__ void wait_gates() {
#ifdef ASYNC_LDS_OK
#if __has_builtin(__builtin_amdgcn_s_wait_asynccnt)
  __builtin_amdgcn_s_wait_asynccnt(0);
#else
  asm volatile("s_wait_asynccnt 0" ::: "memory");
#endif
#endif
}

// ---------------- f32 -> bf16 weight conversion with K padding ----------------
__global__ __launch_bounds__(256) void convert_kernel(const float* __restrict__ src,
                                                      __bf16* __restrict__ dst,
                                                      int rows, int ks, int kd) {
  int idx = blockIdx.x * 256 + threadIdx.x;
  int total = rows * kd;
  if (idx >= total) return;
  int r = idx / kd, k = idx - r * kd;
  float v = (k < ks) ? src[(size_t)r * ks + k] : 0.0f;
  dst[idx] = (__bf16)v;
}

// ---------------- embedding gather + cast + pad ----------------
__global__ __launch_bounds__(256) void embed_kernel(const int* __restrict__ words,
                                                    const float* __restrict__ emb,
                                                    __bf16* __restrict__ x) {
  int idx = blockIdx.x * 256 + threadIdx.x;
  if (idx >= NROW * EMB_PAD) return;
  int row = idx / EMB_PAD, k = idx - row * EMB_PAD;
  float v = (k < EMB) ? emb[(size_t)words[row] * EMB + k] : 0.0f;
  x[idx] = (__bf16)v;
}

// ---------------- big input GEMM: gates = A[NROW,Kpad] @ W^T + b_ih + b_hh ----------------
__global__ __launch_bounds__(256) void input_gemm_kernel(const __bf16* __restrict__ A, int Kpad,
                                                         const __bf16* __restrict__ W,
                                                         const float* __restrict__ bih,
                                                         const float* __restrict__ bhh,
                                                         float* __restrict__ gates) {
  int mt = blockIdx.x;
  int wave = threadIdx.x >> 5, lane = threadIdx.x & 31;
  int row0 = mt * 16;
  v8f acc[8];
#pragma unroll
  for (int j = 0; j < 8; ++j) acc[j] = (v8f)(0.0f);
  int nk = Kpad >> 5;
  for (int kc = 0; kc < nk; ++kc) {
    v16bf a = load_a_frag(A, Kpad, row0, kc * 32, lane);
#pragma unroll
    for (int j = 0; j < 8; ++j) {
      v16bf b = load_b_frag(W, Kpad, (wave * 8 + j) * 16, kc * 32, lane);
      acc[j] = wmma_bf16(a, b, acc[j]);
    }
  }
  int rbase = row0 + ((lane < 16) ? 0 : 8);
  int cl = lane & 15;
#pragma unroll
  for (int j = 0; j < 8; ++j) {
    int col = (wave * 8 + j) * 16 + cl;
    float bias = bih[col] + bhh[col];
#pragma unroll
    for (int v = 0; v < 8; ++v)
      gates[(size_t)(rbase + v) * G4 + col] = acc[j][v] + bias;
  }
}

// ---------------- sequential LSTM recurrence (one block per direction) ----------------
// Double-buffered in LDS: h (2x16KB) and input-gate blocks (2x128KB) streamed by the
// async engine (ASYNCcnt) one step ahead of the WMMA pipeline.
#define LSTM_SMEM (2*BATCH*HID*2 /*h bf16*/ + 2*BATCH*G4*4 /*gates f32*/)   // 294912 B

__global__ __launch_bounds__(256) void lstm_recurrent_kernel(const float* __restrict__ gatesF,
                                                             const float* __restrict__ gatesB,
                                                             const __bf16* __restrict__ whhF,
                                                             const __bf16* __restrict__ whhB,
                                                             __bf16* __restrict__ hcat) {
  int dir = blockIdx.x;                       // 0 = forward, 1 = backward
  const float*  gates = dir ? gatesB : gatesF;
  const __bf16* whh   = dir ? whhB   : whhF;
  int hoff = dir ? HID : 0;
  int wave = threadIdx.x >> 5, lane = threadIdx.x & 31;
  int tid = threadIdx.x;

  extern __shared__ char smem[];
  __bf16* hb0 = (__bf16*)smem;                          // [32][256]
  __bf16* hb1 = hb0 + BATCH * HID;                      // [32][256]
  float*  gl0 = (float*)(smem + 2 * BATCH * HID * 2);   // [32][1024]
  float*  gl1 = gl0 + BATCH * G4;                       // [32][1024]

  for (int i = tid; i < BATCH * HID; i += 256) hb0[i] = (__bf16)0.0f;
  float cst[2][2][8];
#pragma unroll
  for (int mt = 0; mt < 2; ++mt)
#pragma unroll
    for (int jj = 0; jj < 2; ++jj)
#pragma unroll
      for (int v = 0; v < 8; ++v) cst[mt][jj][v] = 0.0f;

  // prologue: stream step-0 gates into buffer 0
  int s0 = dir ? (S_LEN - 1) : 0;
  fetch_gates(gates + (size_t)(s0 * BATCH) * G4, gl0, tid);
  wait_gates();
  __syncthreads();

  int cur = 0;
  int cl = lane & 15;
  int loff = (lane < 16) ? 0 : 8;
  for (int st = 0; st < S_LEN; ++st) {
    int s = dir ? (S_LEN - 1 - st) : st;
    // 1) kick off async stream of next step's gates into the other buffer
    int sn = dir ? (s - 1) : (s + 1);
    if (sn >= 0 && sn < S_LEN)
      fetch_gates(gates + (size_t)(sn * BATCH) * G4, cur ? gl0 : gl1, tid);

    // 2) preload C tiles from current LDS gate buffer (wave w owns n-tiles w+8j)
    const float* gcur = cur ? gl1 : gl0;
    v8f acc[2][8];
#pragma unroll
    for (int mt = 0; mt < 2; ++mt) {
      int rbase = mt * 16 + loff;
#pragma unroll
      for (int j = 0; j < 8; ++j) {
        int col = (wave + 8 * j) * 16 + cl;
        v8f c;
#pragma unroll
        for (int v = 0; v < 8; ++v) c[v] = gcur[(rbase + v) * G4 + col];
        acc[mt][j] = c;
      }
    }
    // 3) recurrent h @ Whh^T : K = 256 (8 chunks of 32)
    const __bf16* hbcur = cur ? hb1 : hb0;
#pragma unroll
    for (int kc = 0; kc < 8; ++kc) {
      v16bf a0 = load_a_frag(hbcur, HID,  0, kc * 32, lane);
      v16bf a1 = load_a_frag(hbcur, HID, 16, kc * 32, lane);
#pragma unroll
      for (int j = 0; j < 8; ++j) {
        v16bf b = load_b_frag(whh, HID, (wave + 8 * j) * 16, kc * 32, lane);
        acc[0][j] = wmma_bf16(a0, b, acc[0][j]);
        acc[1][j] = wmma_bf16(a1, b, acc[1][j]);
      }
    }
    // 4) LSTM cell: i=tiles{w,w+8}, f=+16, g=+32, o=+48 -> acc idx jj, jj+2, jj+4, jj+6
    __bf16* hbnext = cur ? hb0 : hb1;
#pragma unroll
    for (int mt = 0; mt < 2; ++mt) {
#pragma unroll
      for (int jj = 0; jj < 2; ++jj) {
        v8f iv = acc[mt][jj], fv = acc[mt][jj + 2], gv = acc[mt][jj + 4], ov = acc[mt][jj + 6];
        int col = (wave + 8 * jj) * 16 + cl;         // hidden-unit index 0..255
        int m0  = mt * 16 + loff;
#pragma unroll
        for (int v = 0; v < 8; ++v) {
          float cn = sigm(fv[v]) * cst[mt][jj][v] + sigm(iv[v]) * tanhf(gv[v]);
          cst[mt][jj][v] = cn;
          float h = sigm(ov[v]) * tanhf(cn);
          __bf16 hbv = (__bf16)h;
          hbnext[(m0 + v) * HID + col] = hbv;
          hcat[(size_t)(s * BATCH + m0 + v) * IN1 + hoff + col] = hbv;
        }
      }
    }
    // 5) next gates + all LDS traffic complete before flipping buffers
    wait_gates();
    __syncthreads();
    cur ^= 1;
  }
}

// ---------------- FC: em[NROW,32] = h1[NROW,512] @ fc_w^T + fc_b ----------------
__global__ __launch_bounds__(256) void fc_kernel(const __bf16* __restrict__ h1,
                                                 const __bf16* __restrict__ fcw,
                                                 const float* __restrict__ fcb,
                                                 float* __restrict__ em) {
  int wave = threadIdx.x >> 5, lane = threadIdx.x & 31;
  int mt = blockIdx.x * 8 + wave;
  v8f acc[2];
  acc[0] = (v8f)(0.0f); acc[1] = (v8f)(0.0f);
#pragma unroll
  for (int kc = 0; kc < 16; ++kc) {
    v16bf a = load_a_frag(h1, IN1, mt * 16, kc * 32, lane);
#pragma unroll
    for (int j = 0; j < 2; ++j) {
      v16bf b = load_b_frag(fcw, IN1, j * 16, kc * 32, lane);
      acc[j] = wmma_bf16(a, b, acc[j]);
    }
  }
  int rbase = mt * 16 + ((lane < 16) ? 0 : 8);
  int cl = lane & 15;
#pragma unroll
  for (int j = 0; j < 2; ++j) {
    int col = j * 16 + cl;
    float bias = fcb[col];
#pragma unroll
    for (int v = 0; v < 8; ++v)
      em[(size_t)(rbase + v) * TAGS + col] = acc[j][v] + bias;
  }
}

// ---------------- CRF loss: one wave per batch element, one tag per lane ----------------
__global__ __launch_bounds__(1024) void crf_loss_kernel(const float* __restrict__ em,
                                                        const int* __restrict__ tags,
                                                        const float* __restrict__ start,
                                                        const float* __restrict__ trans,
                                                        const float* __restrict__ end,
                                                        float* __restrict__ loss_out) {
  int b = threadIdx.x >> 5, t = threadIdx.x & 31;
  __shared__ float trs[TAGS * TAGS];
  __shared__ float part[BATCH];
  for (int i = threadIdx.x; i < TAGS * TAGS; i += 1024) trs[i] = trans[i];
  __syncthreads();

  float alpha = start[t] + em[(size_t)(0 * BATCH + b) * TAGS + t];
  for (int s = 1; s < S_LEN; ++s) {
    float e = em[(size_t)(s * BATCH + b) * TAGS + t];
    float mx = -1e30f, sm = 0.0f;
#pragma unroll 4
    for (int j = 0; j < TAGS; ++j) {
      float aj = __shfl(alpha, j, 32);
      float v = aj + trs[j * TAGS + t];
      if (v > mx) { sm = sm * __expf(mx - v) + 1.0f; mx = v; }
      else        { sm += __expf(v - mx); }
    }
    alpha = mx + __logf(sm) + e;
  }
  float v = alpha + end[t];
  float mx = v;
  for (int o = 16; o; o >>= 1) mx = fmaxf(mx, __shfl_xor(mx, o, 32));
  float sm = __expf(v - mx);
  for (int o = 16; o; o >>= 1) sm += __shfl_xor(sm, o, 32);
  float logZ = mx + __logf(sm);
  float num = 0.0f;
  for (int s = t; s < S_LEN; s += 32) {
    int tg = tags[s * BATCH + b];
    num += em[(size_t)(s * BATCH + b) * TAGS + tg];
    if (s > 0) num += trs[tags[(s - 1) * BATCH + b] * TAGS + tg];
  }
  if (t == 0) num += start[tags[b]] + end[tags[(S_LEN - 1) * BATCH + b]];
  for (int o = 16; o; o >>= 1) num += __shfl_xor(num, o, 32);
  if (t == 0) part[b] = logZ - num;
  __syncthreads();
  if (threadIdx.x == 0) {
    float tot = 0.0f;
    for (int i = 0; i < BATCH; ++i) tot += part[i];
    *loss_out = tot;
  }
}

// ---------------- Viterbi: one wave per batch, backptrs to ws, lane 0 backtracks ----------------
__global__ __launch_bounds__(1024) void viterbi_kernel(const float* __restrict__ em,
                                                       const float* __restrict__ start,
                                                       const float* __restrict__ trans,
                                                       const float* __restrict__ end,
                                                       int* __restrict__ bp,
                                                       float* __restrict__ path_out) {
  int b = threadIdx.x >> 5, t = threadIdx.x & 31;
  __shared__ float trs[TAGS * TAGS];
  for (int i = threadIdx.x; i < TAGS * TAGS; i += 1024) trs[i] = trans[i];
  __syncthreads();

  float score = start[t] + em[(size_t)(0 * BATCH + b) * TAGS + t];
  for (int s = 1; s < S_LEN; ++s) {
    float mx = -1e30f; int arg = 0;
#pragma unroll 4
    for (int j = 0; j < TAGS; ++j) {
      float v = __shfl(score, j, 32) + trs[j * TAGS + t];
      if (v > mx) { mx = v; arg = j; }
    }
    bp[((s - 1) * BATCH + b) * TAGS + t] = arg;
    score = mx + em[(size_t)(s * BATCH + b) * TAGS + t];
  }
  float v = score + end[t]; int bi = t;
  for (int o = 16; o; o >>= 1) {
    float ov = __shfl_xor(v, o, 32); int oi = __shfl_xor(bi, o, 32);
    if (ov > v || (ov == v && oi < bi)) { v = ov; bi = oi; }
  }
  if (t == 0) {
    int tag = bi;
    path_out[(S_LEN - 1) * BATCH + b] = (float)tag;
    for (int s = S_LEN - 2; s >= 0; --s) {
      tag = bp[(s * BATCH + b) * TAGS + tag];
      path_out[s * BATCH + b] = (float)tag;
    }
  }
}

static inline int cdiv(int a, int b) { return (a + b - 1) / b; }

extern "C" void kernel_launch(void* const* d_in, const int* in_sizes, int n_in,
                              void* d_out, int out_size, void* d_ws, size_t ws_size,
                              hipStream_t stream) {
  const int*   words     = (const int*)  d_in[0];
  const int*   tags      = (const int*)  d_in[2];
  const float* emb_table = (const float*)d_in[3];
  const float* fc_w      = (const float*)d_in[4];
  const float* fc_b      = (const float*)d_in[5];
  const float* crf_start = (const float*)d_in[6];
  const float* crf_end   = (const float*)d_in[7];
  const float* crf_trans = (const float*)d_in[8];
  const float* w_ih_l0f = (const float*)d_in[9];  const float* w_hh_l0f = (const float*)d_in[10];
  const float* b_ih_l0f = (const float*)d_in[11]; const float* b_hh_l0f = (const float*)d_in[12];
  const float* w_ih_l0b = (const float*)d_in[13]; const float* w_hh_l0b = (const float*)d_in[14];
  const float* b_ih_l0b = (const float*)d_in[15]; const float* b_hh_l0b = (const float*)d_in[16];
  const float* w_ih_l1f = (const float*)d_in[17]; const float* w_hh_l1f = (const float*)d_in[18];
  const float* b_ih_l1f = (const float*)d_in[19]; const float* b_hh_l1f = (const float*)d_in[20];
  const float* w_ih_l1b = (const float*)d_in[21]; const float* w_hh_l1b = (const float*)d_in[22];
  const float* b_ih_l1b = (const float*)d_in[23]; const float* b_hh_l1b = (const float*)d_in[24];

  size_t off = 0;
  auto carve = [&](size_t bytes) -> void* {
    void* p = (char*)d_ws + off;
    off += (bytes + 255) & ~(size_t)255;
    return p;
  };
  __bf16* xbf    = (__bf16*)carve((size_t)NROW * EMB_PAD * 2);
  __bf16* wih0f  = (__bf16*)carve((size_t)G4 * EMB_PAD * 2);
  __bf16* wih0b  = (__bf16*)carve((size_t)G4 * EMB_PAD * 2);
  __bf16* whh0f  = (__bf16*)carve((size_t)G4 * HID * 2);
  __bf16* whh0b  = (__bf16*)carve((size_t)G4 * HID * 2);
  __bf16* wih1f  = (__bf16*)carve((size_t)G4 * IN1 * 2);
  __bf16* wih1b  = (__bf16*)carve((size_t)G4 * IN1 * 2);
  __bf16* whh1f  = (__bf16*)carve((size_t)G4 * HID * 2);
  __bf16* whh1b  = (__bf16*)carve((size_t)G4 * HID * 2);
  __bf16* fcwbf  = (__bf16*)carve((size_t)TAGS * IN1 * 2);
  float*  gatesF = (float*) carve((size_t)NROW * G4 * 4);   // reused by layer 0 & 1
  float*  gatesB = (float*) carve((size_t)NROW * G4 * 4);
  __bf16* h0     = (__bf16*)carve((size_t)NROW * IN1 * 2);
  __bf16* h1     = (__bf16*)carve((size_t)NROW * IN1 * 2);
  float*  em     = (float*) carve((size_t)NROW * TAGS * 4);
  int*    bp     = (int*)   carve((size_t)S_LEN * BATCH * TAGS * 4);
  (void)ws_size; (void)in_sizes; (void)n_in; (void)out_size;

  float* path_out = (float*)d_out;          // [S,B] viterbi path as float
  float* loss_out = (float*)d_out + NROW;   // scalar loss

  auto conv = [&](const float* s, __bf16* d, int rows, int ks, int kd) {
    int tot = rows * kd;
    convert_kernel<<<cdiv(tot, 256), 256, 0, stream>>>(s, d, rows, ks, kd);
  };
  conv(w_ih_l0f, wih0f, G4, EMB, EMB_PAD);
  conv(w_ih_l0b, wih0b, G4, EMB, EMB_PAD);
  conv(w_hh_l0f, whh0f, G4, HID, HID);
  conv(w_hh_l0b, whh0b, G4, HID, HID);
  conv(w_ih_l1f, wih1f, G4, IN1, IN1);
  conv(w_ih_l1b, wih1b, G4, IN1, IN1);
  conv(w_hh_l1f, whh1f, G4, HID, HID);
  conv(w_hh_l1b, whh1b, G4, HID, HID);
  conv(fc_w,     fcwbf, TAGS, IN1, IN1);

  embed_kernel<<<cdiv(NROW * EMB_PAD, 256), 256, 0, stream>>>(words, emb_table, xbf);

  input_gemm_kernel<<<NROW / 16, 256, 0, stream>>>(xbf, EMB_PAD, wih0f, b_ih_l0f, b_hh_l0f, gatesF);
  input_gemm_kernel<<<NROW / 16, 256, 0, stream>>>(xbf, EMB_PAD, wih0b, b_ih_l0b, b_hh_l0b, gatesB);
  lstm_recurrent_kernel<<<2, 256, LSTM_SMEM, stream>>>(gatesF, gatesB, whh0f, whh0b, h0);

  input_gemm_kernel<<<NROW / 16, 256, 0, stream>>>(h0, IN1, wih1f, b_ih_l1f, b_hh_l1f, gatesF);
  input_gemm_kernel<<<NROW / 16, 256, 0, stream>>>(h0, IN1, wih1b, b_ih_l1b, b_hh_l1b, gatesB);
  lstm_recurrent_kernel<<<2, 256, LSTM_SMEM, stream>>>(gatesF, gatesB, whh1f, whh1b, h1);

  fc_kernel<<<NROW / 16 / 8, 256, 0, stream>>>(h1, fcwbf, fc_b, em);

  crf_loss_kernel<<<1, 1024, 0, stream>>>(em, tags, crf_start, crf_trans, crf_end, loss_out);
  viterbi_kernel<<<1, 1024, 0, stream>>>(em, crf_start, crf_trans, crf_end, bp, path_out);
}
